// FullAttention_1340029796603
// MI455X (gfx1250) — matrix operand order repaired and checked by reading, not compiled, not run on hardware
//
#include <hip/hip_runtime.h>

// ---------------------------------------------------------------------------
// Causal flash attention forward for MI455X (gfx1250), wave32 + WMMA bf16.
// B=4, L=S=2048, H=16, E=64, fp32 in/out, bf16 matmuls, fp32 accumulate.
// Pipelined KV staging; exp2-domain online softmax (DPP16 max reduction,
// WMMA row sums); key-interleaved P/V layout for packed b32 P stores.
// ---------------------------------------------------------------------------

typedef __bf16 v16bf  __attribute__((ext_vector_type(16)));
typedef __bf16 bf16x8 __attribute__((ext_vector_type(8)));
typedef __bf16 bf16x4 __attribute__((ext_vector_type(4)));
typedef __bf16 bf16x2 __attribute__((ext_vector_type(2)));
typedef float  v8f    __attribute__((ext_vector_type(8)));

constexpr int B_ = 4, L_ = 2048, S_ = 2048, H_ = 16, E_ = 64;
constexpr int BM = 64;   // query rows per workgroup (4 waves x 16)
constexpr int BN = 32;   // keys per KV tile
// Fold softmax scale and log2(e) into Q so scores are already in exp2 domain.
constexpr float QSCALE = 0.125f * 1.44269504088896340736f; // 1/sqrt(64)*log2(e)
constexpr float NEGINF = -1e30f;

__device__ __forceinline__ __bf16 f2bf(float f) { return (__bf16)f; }

// Max across each 16-lane group using DPP16 only (VALU pipe, no LDS).
__device__ __forceinline__ float rowmax16(float x) {
#define DPP_MAX_STEP(ctrl)                                                    \
  {                                                                           \
    int t_ = __builtin_amdgcn_update_dpp(0, __builtin_bit_cast(int, x),       \
                                         (ctrl), 0xF, 0xF, true);             \
    x = fmaxf(x, __builtin_bit_cast(float, t_));                              \
  }
  DPP_MAX_STEP(0xB1);   // quad_perm [1,0,3,2] : xor 1
  DPP_MAX_STEP(0x4E);   // quad_perm [2,3,0,1] : xor 2
  DPP_MAX_STEP(0x141);  // ROW_HALF_MIRROR     : fold 8s
  DPP_MAX_STEP(0x140);  // ROW_MIRROR          : fold 16s
#undef DPP_MAX_STEP
  return x;
}

// One 32-key tile: scores (4 WMMA), online softmax, P relayout,
// row sums (1 WMMA), P.V (4 WMMA).
// Contraction axis of P/V uses interleaved key order pi(2j)=j, pi(2j+1)=16+j,
// so each row's (pl, pr) pair is one contiguous b32 LDS store.
template <bool MASKED>
__device__ __forceinline__ void tile_compute(
    int kt, int qrow_base, int half, int ln,
    const v16bf& a0, const v16bf& a1, const v16bf& bones,
    v8f& o0, v8f& o1, v8f& o2, v8f& o3,
    float (&mrow)[8], float (&lrow)[8],
    const __bf16* __restrict__ Kt, const __bf16* __restrict__ Vt,
    __bf16* __restrict__ Pwv) {
  // B operand here (32x16, contiguous k order, same as A): lane = N + 16*(K/16), elem = K%16.
  v16bf bk0L = *(const v16bf*)&Kt[ ln       * E_ +      16 * half];
  v16bf bk1L = *(const v16bf*)&Kt[ ln       * E_ + 32 + 16 * half];
  v16bf bk0R = *(const v16bf*)&Kt[(16 + ln) * E_ +      16 * half];
  v16bf bk1R = *(const v16bf*)&Kt[(16 + ln) * E_ + 32 + 16 * half];

  v8f sL = {}, sR = {};
  sL = __builtin_amdgcn_wmma_f32_16x16x32_bf16(false, a0, false, bk0L, (short)0, sL, false, false);
  sL = __builtin_amdgcn_wmma_f32_16x16x32_bf16(false, a1, false, bk1L, (short)0, sL, false, false);
  sR = __builtin_amdgcn_wmma_f32_16x16x32_bf16(false, a0, false, bk0R, (short)0, sR, false, false);
  sR = __builtin_amdgcn_wmma_f32_16x16x32_bf16(false, a1, false, bk1R, (short)0, sR, false, false);

  // Online softmax; C layout: row = i + 8*half, col = ln (original keys).
#pragma unroll
  for (int i = 0; i < 8; ++i) {
    float sl = sL[i];
    float sr = sR[i];
    if (MASKED) {
      const int q = qrow_base + 8 * half + i;
      sl = (kt + ln      <= q) ? sl : NEGINF;
      sr = (kt + 16 + ln <= q) ? sr : NEGINF;
    }
    const float tm = rowmax16(fmaxf(sl, sr));
    const float mnew = fmaxf(mrow[i], tm);
    const float corr = exp2f(mrow[i] - mnew);
    const float pl = exp2f(sl - mnew);
    const float pr = exp2f(sr - mnew);
    mrow[i] = mnew;
    lrow[i] *= corr;                       // row sum added after WMMA below
    o0[i] *= corr; o1[i] *= corr; o2[i] *= corr; o3[i] *= corr;
    const int r = 8 * half + i;
    bf16x2 pp; pp[0] = f2bf(pl); pp[1] = f2bf(pr);
    *(bf16x2*)&Pwv[r * BN + 2 * ln] = pp;  // keys (ln, 16+ln) -> slots 2ln,2ln+1
  }

  // Re-read P in A-matrix layout (16x32 over interleaved key order).
  v16bf pa;
  {
    bf16x8 p0 = *(const bf16x8*)&Pwv[ln * BN +     16 * half];
    bf16x8 p1 = *(const bf16x8*)&Pwv[ln * BN + 8 + 16 * half];
#pragma unroll
    for (int e = 0; e < 8; ++e) { pa[e] = p0[e]; pa[e + 8] = p1[e]; }
  }

  // Row sums of P via WMMA against an all-ones B (permutation invariant).
  v8f ls = {};
  ls = __builtin_amdgcn_wmma_f32_16x16x32_bf16(false, pa, false, bones, (short)0, ls, false, false);
#pragma unroll
  for (int i = 0; i < 8; ++i) lrow[i] += ls[i];

  // O += P.V  (Vt columns already stored in the interleaved key order).
  v16bf bv0 = *(const v16bf*)&Vt[( 0 + ln) * BN + 16 * half];
  v16bf bv1 = *(const v16bf*)&Vt[(16 + ln) * BN + 16 * half];
  v16bf bv2 = *(const v16bf*)&Vt[(32 + ln) * BN + 16 * half];
  v16bf bv3 = *(const v16bf*)&Vt[(48 + ln) * BN + 16 * half];
  o0 = __builtin_amdgcn_wmma_f32_16x16x32_bf16(false, pa, false, bv0, (short)0, o0, false, false);
  o1 = __builtin_amdgcn_wmma_f32_16x16x32_bf16(false, pa, false, bv1, (short)0, o1, false, false);
  o2 = __builtin_amdgcn_wmma_f32_16x16x32_bf16(false, pa, false, bv2, (short)0, o2, false, false);
  o3 = __builtin_amdgcn_wmma_f32_16x16x32_bf16(false, pa, false, bv3, (short)0, o3, false, false);
}

__global__ __launch_bounds__(128)
void fa_causal_fwd(const float* __restrict__ Qg, const float* __restrict__ Kg,
                   const float* __restrict__ Vg, float* __restrict__ Og) {
  __shared__ __align__(16) __bf16 Kt[BN * E_];      // [key][dim]     4 KB
  __shared__ __align__(16) __bf16 Vt[E_ * BN];      // [dim][perm key] 4 KB
  __shared__ __align__(16) __bf16 Pw[4][16 * BN];   // per-wave P     4 KB

  const int tid  = threadIdx.x;
  const int wv   = tid >> 5;
  const int lane = tid & 31;
  const int half = lane >> 4;
  const int ln   = lane & 15;

  const int q0 = blockIdx.x * BM;
  const int bh = blockIdx.y;
  const int b  = bh / H_;
  const int h  = bh % H_;
  const int qrow_base = q0 + wv * 16;

  // ---- Q tile -> two 16x32 bf16 A-matrices, with QSCALE folded in ----
  const float* qptr = Qg + (((size_t)b * L_ + (qrow_base + ln)) * H_ + h) * E_;
  v16bf a0, a1;
#pragma unroll
  for (int e = 0; e < 8; ++e) {
    a0[e]     = f2bf(qptr[      16 * half + e] * QSCALE);
    a0[e + 8] = f2bf(qptr[ 8 + 16 * half + e] * QSCALE);
    a1[e]     = f2bf(qptr[32 + 16 * half + e] * QSCALE);
    a1[e + 8] = f2bf(qptr[40 + 16 * half + e] * QSCALE);
  }

  v16bf bones;
#pragma unroll
  for (int e = 0; e < 16; ++e) bones[e] = (__bf16)1.0f;

  v8f o0 = {}, o1 = {}, o2 = {}, o3 = {};
  float mrow[8], lrow[8];
#pragma unroll
  for (int i = 0; i < 8; ++i) { mrow[i] = NEGINF; lrow[i] = 0.f; }

  const int kend = (q0 + BM < S_) ? (q0 + BM) : S_;

  // ---- Pipelined KV staging: 128 threads -> 32 keys x 4 chunks of 16 ----
  const int skey = tid >> 2;
  const int sch  = (tid & 3) * 16;
  // Interleaved key-column for V: pi^-1(k) = k<16 ? 2k : 2(k-16)+1.
  const int vcol = (skey < 16) ? (2 * skey) : (2 * (skey - 16) + 1);
  const size_t kvstride = (size_t)H_ * E_;

  float4 kreg[4], vreg[4];
  auto load_tile = [&](int kt) {
    const size_t gidx = (((size_t)b * S_ + (kt + skey)) * H_ + h) * E_ + sch;
    const float* kr = Kg + gidx;
    const float* vr = Vg + gidx;
#pragma unroll
    for (int c = 0; c < 4; ++c) {
      kreg[c] = *(const float4*)(kr + 4 * c);
      vreg[c] = *(const float4*)(vr + 4 * c);
    }
    // L2 prefetch two tiles ahead of compute.
    __builtin_prefetch(kr + BN * kvstride, 0, 1);
    __builtin_prefetch(vr + BN * kvstride, 0, 1);
  };
  auto store_tile = [&]() {
#pragma unroll
    for (int c = 0; c < 4; ++c) {
      const float kf[4] = {kreg[c].x, kreg[c].y, kreg[c].z, kreg[c].w};
      const float vf[4] = {vreg[c].x, vreg[c].y, vreg[c].z, vreg[c].w};
      bf16x4 kp;
#pragma unroll
      for (int e = 0; e < 4; ++e) {
        kp[e] = f2bf(kf[e]);
        Vt[(sch + 4 * c + e) * BN + vcol] = f2bf(vf[e]);   // [dim][perm key]
      }
      *(bf16x4*)&Kt[skey * E_ + sch + 4 * c] = kp;         // b64 store
    }
  };

  load_tile(0);
  for (int kt = 0; kt < kend; kt += BN) {
    __syncthreads();        // previous tile fully consumed
    store_tile();
    __syncthreads();        // tile visible to all waves
    if (kt + BN < kend) load_tile(kt + BN);  // overlap loads with compute
    if (kt + BN <= q0) {    // strictly below the diagonal block: no masking
      tile_compute<false>(kt, qrow_base, half, ln, a0, a1, bones,
                          o0, o1, o2, o3, mrow, lrow, Kt, Vt, Pw[wv]);
    } else {                // diagonal tiles: per-element causal mask
      tile_compute<true>(kt, qrow_base, half, ln, a0, a1, bones,
                         o0, o1, o2, o3, mrow, lrow, Kt, Vt, Pw[wv]);
    }
  }

  // ---- Epilogue: O / l, scatter to [B, L, H, E] ----
#pragma unroll
  for (int i = 0; i < 8; ++i) {
    const int q = qrow_base + 8 * half + i;
    const float inv = 1.0f / lrow[i];
    float* orow = Og + (((size_t)b * L_ + q) * H_ + h) * E_;
    orow[ 0 + ln] = o0[i] * inv;
    orow[16 + ln] = o1[i] * inv;
    orow[32 + ln] = o2[i] * inv;
    orow[48 + ln] = o3[i] * inv;
  }
}

extern "C" void kernel_launch(void* const* d_in, const int* in_sizes, int n_in,
                              void* d_out, int out_size, void* d_ws, size_t ws_size,
                              hipStream_t stream) {
  (void)in_sizes; (void)n_in; (void)d_ws; (void)ws_size; (void)out_size;
  const float* Q = (const float*)d_in[0];
  const float* K = (const float*)d_in[1];
  const float* V = (const float*)d_in[2];
  // d_in[3] (causal mask) is deterministic triu; recomputed in-kernel.
  float* O = (float*)d_out;
  dim3 grid(L_ / BM, B_ * H_);
  dim3 block(128);
  hipLaunchKernelGGL(fa_causal_fwd, grid, block, 0, stream, Q, K, V, O);
}
